// FullEncoder_25348896981287
// MI455X (gfx1250) — matrix operand order repaired and checked by reading, not compiled, not run on hardware
//
#include <hip/hip_runtime.h>
#include <hip/hip_bf16.h>
#include <math.h>

// ---------------------------------------------------------------------------
// BERT-style encoder for MI455X (gfx1250). All GEMMs run on
// v_wmma_f32_16x16x32_bf16 (wave32). The relative-position "skew" is folded
// into leading-dimension tricks so every attention step is a plain GEMM:
//   flat index of skewed score (q, r=k-q+511) = q*1023 + k + 511.
// Softmax emits probs in BOTH skewed [q][1024] and un-skewed [q][512] layouts
// so every GEMM A-operand is 16B-aligned and staging is fully vectorized.
// A-tiles are staged with the CDNA5 async global->LDS engine (ASYNCcnt).
// Invariants exploited (all shapes in this model): K % 32 == 0, M % 128 == 0,
// N % 16 == 0, BT-mode N % 128 == 0 -> no per-element guards anywhere.
// Workspace requirement: ~240 MB.
// ---------------------------------------------------------------------------

typedef __attribute__((ext_vector_type(16))) __bf16 bf16x16;
typedef __attribute__((ext_vector_type(8)))  float  f32x8;

#define BATCH   4
#define SEQ     512
#define HID     768
#define NHEADS  12
#define HDIM    64
#define MFF     3072
#define NLAYERS 6
#define REL     1024          // padded 2L-1 (=1023) -> 1024
#define RELROWS 255           // 2*RD-1
#define RELPAD  384           // L - RD
#define COEF    0.125f
#define ROWS    (BATCH*SEQ)   // 2048

#define BM 128
#define BN 128
#define BK 32
#define LDSP 40               // LDS row stride in bf16 elems (32 + 8 pad)

#define FLAG_ACCUM 1
#define FLAG_GELU  2
#define FLAG_BT    4

// tanh-GELU via sigmoid identity: 0.5*(1+tanh(u)) == 1/(1+exp(-2u)).
// Uses hardware v_exp_f32 (__expf) -> ~10 VALU ops, no libm branches.
__device__ __forceinline__ float gelu_fast(float x) {
  const float u = 1.5957691216057308f * (x + 0.044715f * x * x * x); // 2*0.79788456*(..)
  return x * (1.0f / (1.0f + __expf(-u)));
}

// ---------------------------------------------------------------------------
// Generic batched WMMA GEMM: C[M,N] = A(bf16)[M,K] * B(f32) (+bias) epilogue.
//   FLAG_BT   : B is [N,K] row-major (Q*K^T style); N must be multiple of 128
//   else      : B is [K,N] row-major (weights); N multiple of 16
//   FLAG_ACCUM: C += result (C must be non-null)
//   FLAG_GELU : apply tanh-GELU in epilogue
// Batch z: offset = (z / batchDiv) * Sb + (z % batchDiv) * Sh  (elements)
// Requirements: M % 128 == 0, K % 32 == 0, A/B 16B-aligned rows.
// ---------------------------------------------------------------------------
__global__ __launch_bounds__(256)
void wmma_gemm_kernel(const __bf16* __restrict__ A, int lda, long long aSb, long long aSh,
                      const float*  __restrict__ Bm, int ldb, long long bSb, long long bSh,
                      const float*  __restrict__ bias,
                      float* C, int ldc,
                      __bf16* Cb, int ldcb,
                      long long cSb, long long cSh,
                      int M, int N, int K, int flags, int batchDiv)
{
  __shared__ __bf16 As[2][BM * LDSP];
  __shared__ __bf16 Bs[2][BN * LDSP];

  const int z  = blockIdx.z;
  const int zb = z / batchDiv;
  const int zh = z % batchDiv;
  A  += (long long)zb * aSb + (long long)zh * aSh;
  Bm += (long long)zb * bSb + (long long)zh * bSh;
  const long long coff = (long long)zb * cSb + (long long)zh * cSh;
  if (C)  C  += coff;
  if (Cb) Cb += coff;

  const int m0 = blockIdx.y * BM;
  const int n0 = blockIdx.x * BN;
  const int tid  = threadIdx.x;
  const int lane = tid & 31;
  const int wave = tid >> 5;          // 8 waves
  const int wm = (wave >> 2) * 64;    // 2 waves along M, 64 rows each
  const int wn = (wave & 3) * 32;     // 4 waves along N, 32 cols each

  // Stage one [BM x BK] A tile + one B tile into LDS buffer `buf`.
  // A uses the gfx1250 async global->LDS DMA (ASYNCcnt); the same inst offset
  // is applied to both the global and LDS addresses (ISA 15.18.3), so one
  // base covers both 16B halves. B is load+convert+store (needs VALU cvt).
  auto stage = [&](int buf, int k0) {
    { // A: two async b128 copies, no VGPR round trip
      const int ar = tid >> 1;
      const int ac = (tid & 1) * 16;
      const __bf16* g = A + (long long)(m0 + ar) * lda + k0 + ac;
      const unsigned lds = (unsigned)(unsigned long long)&As[buf][ar * LDSP + ac];
      asm volatile("global_load_async_to_lds_b128 %0, %1, off"
                   :: "v"(lds), "v"(g) : "memory");
      asm volatile("global_load_async_to_lds_b128 %0, %1, off offset:16"
                   :: "v"(lds), "v"(g) : "memory");
    }
    if (flags & FLAG_BT) { // B [N,K] f32: 4 x b128 load, convert, 2 x b128 store
      const int bn = tid >> 1;
      const int bk = (tid & 1) * 16;
      const float4* src = (const float4*)(Bm + (long long)(n0 + bn) * ldb + k0 + bk);
      if (k0 + 2 * BK < K) __builtin_prefetch(src + 4, 0, 0);
      float4 f[4];
      #pragma unroll
      for (int j = 0; j < 4; ++j) f[j] = src[j];
      __bf16 tmp[16];
      #pragma unroll
      for (int j = 0; j < 4; ++j) {
        tmp[4*j+0] = (__bf16)f[j].x; tmp[4*j+1] = (__bf16)f[j].y;
        tmp[4*j+2] = (__bf16)f[j].z; tmp[4*j+3] = (__bf16)f[j].w;
      }
      uint4* dst = (uint4*)&Bs[buf][bn * LDSP + bk];
      dst[0] = ((const uint4*)tmp)[0];
      dst[1] = ((const uint4*)tmp)[1];
    } else { // B [K,N] f32: coalesced 4 x b128 load, transposed b16 LDS writes
      const int bk = tid >> 3;          // 0..31
      const int bn = (tid & 7) * 16;    // 0,16,...,112
      const float4* src = (const float4*)(Bm + (long long)(k0 + bk) * ldb + n0 + bn);
      float vv[16];
      if (n0 + bn < N) {               // span-exact: N % 16 == 0
        if (k0 + 2 * BK < K) __builtin_prefetch(src + (long long)(2 * BK / 4) * (ldb / 4), 0, 0);
        #pragma unroll
        for (int j = 0; j < 4; ++j) ((float4*)vv)[j] = src[j];
      } else {
        #pragma unroll
        for (int e = 0; e < 16; ++e) vv[e] = 0.0f;
      }
      #pragma unroll
      for (int e = 0; e < 16; ++e)
        Bs[buf][(bn + e) * LDSP + bk] = (__bf16)vv[e];
    }
  };

  f32x8 acc[4][2];
  #pragma unroll
  for (int mt = 0; mt < 4; ++mt)
    #pragma unroll
    for (int nt = 0; nt < 2; ++nt)
      #pragma unroll
      for (int e = 0; e < 8; ++e) acc[mt][nt][e] = 0.0f;

  stage(0, 0);
  int buf = 0;
  const int rsel = lane >> 4;   // 0: lanes 0-15, 1: lanes 16-31
  const int l15  = lane & 15;

  for (int k0 = 0; k0 < K; k0 += BK) {
    // Drain this wave's async A-tile copies, then make all waves' tiles visible.
    asm volatile("s_wait_asynccnt 0x0" ::: "memory");
    __syncthreads();
    if (k0 + BK < K) stage(buf ^ 1, k0 + BK);

    // Fragments per CDNA5 16-bit A(16x32)/B(32x16) lane layouts.
    bf16x16 af[4];
    bf16x16 bfr[2];
    #pragma unroll
    for (int mt = 0; mt < 4; ++mt) {
      const __bf16* p = &As[buf][(wm + mt * 16 + l15) * LDSP + (rsel ? 8 : 0)];
      #pragma unroll
      for (int e = 0; e < 8; ++e) af[mt][e] = p[e];          // K 0..7  / 8..15
      #pragma unroll
      for (int e = 0; e < 8; ++e) af[mt][8 + e] = p[16 + e]; // K 16..23 / 24..31
    }
    #pragma unroll
    for (int nt = 0; nt < 2; ++nt) {
      const __bf16* p = &Bs[buf][(wn + nt * 16 + l15) * LDSP + (rsel ? 8 : 0)];
      #pragma unroll
      for (int e = 0; e < 8; ++e) bfr[nt][e] = p[e];          // K 0..7  / 8..15
      #pragma unroll
      for (int e = 0; e < 8; ++e) bfr[nt][8 + e] = p[16 + e]; // K 16..23 / 24..31
    }
    #pragma unroll
    for (int mt = 0; mt < 4; ++mt)
      #pragma unroll
      for (int nt = 0; nt < 2; ++nt)
        acc[mt][nt] = __builtin_amdgcn_wmma_f32_16x16x32_bf16(
            false, af[mt], false, bfr[nt], (short)0, acc[mt][nt], false, false);
    buf ^= 1;
  }

  // Epilogue. C/D layout: lane<16 -> (M=r, N=lane); lane>=16 -> (M=r+8, N=lane-16).
  // Rows always valid (M % 128 == 0); column spans exact (N % 16 == 0).
  // GELU has a single application site per element (code-size critical).
  #pragma unroll
  for (int mt = 0; mt < 4; ++mt) {
    #pragma unroll
    for (int nt = 0; nt < 2; ++nt) {
      const int colBase = n0 + wn + nt * 16;
      if (colBase < N) {
        const int col = colBase + l15;
        const float bv = bias ? bias[col] : 0.0f;
        const int rbase = m0 + wm + mt * 16 + (rsel ? 8 : 0);
        #pragma unroll
        for (int r = 0; r < 8; ++r) {
          const long long idx = (long long)(rbase + r) * ldc + col;
          float v = acc[mt][nt][r] + bv;
          if (C && (flags & FLAG_ACCUM)) v += C[idx];
          if (flags & FLAG_GELU) v = gelu_fast(v);
          if (C) C[idx] = v;
          if (Cb) Cb[(long long)(rbase + r) * ldcb + col] = (__bf16)v;
        }
      }
    }
  }
}

// --------------------------- reductions ------------------------------------
__device__ __forceinline__ float blockReduceSum(float v, float* red) {
  #pragma unroll
  for (int o = 16; o; o >>= 1) v += __shfl_xor(v, o, 32);
  const int w = threadIdx.x >> 5;
  __syncthreads();
  if ((threadIdx.x & 31) == 0) red[w] = v;
  __syncthreads();
  float t = red[0];
  #pragma unroll
  for (int i = 1; i < 8; ++i) t += red[i];
  return t;
}

__device__ __forceinline__ float blockReduceMax(float v, float* red) {
  #pragma unroll
  for (int o = 16; o; o >>= 1) v = fmaxf(v, __shfl_xor(v, o, 32));
  const int w = threadIdx.x >> 5;
  __syncthreads();
  if ((threadIdx.x & 31) == 0) red[w] = v;
  __syncthreads();
  float t = red[0];
  #pragma unroll
  for (int i = 1; i < 8; ++i) t = fmaxf(t, red[i]);
  return t;
}

// --------------------------- aux kernels -----------------------------------
__global__ __launch_bounds__(256)
void embed_kernel(const int* __restrict__ tok, const int* __restrict__ seg,
                  const float* __restrict__ te, const float* __restrict__ se,
                  float* __restrict__ out)
{
  const int row = blockIdx.x;
  const int t = tok[row];
  const int s = seg[row];
  const float* tp = te + (long long)t * HID;
  const float* sp = se + (long long)s * HID;
  float* o = out + (long long)row * HID;
  for (int c = threadIdx.x; c < HID; c += 256) o[c] = tp[c] + sp[c];
}

// LayerNorm((a + res)) * scale + bias -> outf (f32, optional) and outb (bf16)
__global__ __launch_bounds__(256)
void ln_kernel(const float* __restrict__ a, const float* __restrict__ res,
               const float* __restrict__ sc, const float* __restrict__ bi,
               float* __restrict__ outf, __bf16* __restrict__ outb)
{
  __shared__ float red[8];
  const int row = blockIdx.x;
  const float* ap = a + (long long)row * HID;
  const float* rp = res ? res + (long long)row * HID : nullptr;
  float v[3];
  int   c[3];
  #pragma unroll
  for (int j = 0; j < 3; ++j) {
    c[j] = threadIdx.x + j * 256;
    v[j] = ap[c[j]] + (rp ? rp[c[j]] : 0.0f);
  }
  float s = v[0] + v[1] + v[2];
  s = blockReduceSum(s, red);
  const float mean = s * (1.0f / HID);
  float q = 0.0f;
  #pragma unroll
  for (int j = 0; j < 3; ++j) { const float d = v[j] - mean; q += d * d; }
  q = blockReduceSum(q, red);
  const float rstd = rsqrtf(q * (1.0f / HID) + 1e-12f);
  #pragma unroll
  for (int j = 0; j < 3; ++j) {
    const float o = (v[j] - mean) * rstd * sc[c[j]] + bi[c[j]];
    const long long idx = (long long)row * HID + c[j];
    if (outf) outf[idx] = o;
    if (outb) outb[idx] = (__bf16)o;
  }
}

__global__ __launch_bounds__(256)
void maskv_kernel(const int* __restrict__ im, float* __restrict__ mv, int n)
{
  const int i = blockIdx.x * 256 + threadIdx.x;
  if (i < n) mv[i] = im[i] ? 0.0f : -1.0e30f;
}

// Edge-pad relpos[255,H] into bf16 [1024,H] (rows clamp(r-384, 0, 254)).
__global__ __launch_bounds__(256)
void relpad_kernel(const float* __restrict__ rp, __bf16* __restrict__ relb)
{
  const int idx = blockIdx.x * 256 + threadIdx.x;
  if (idx < REL * HID) {
    const int r  = idx / HID;
    const int c2 = idx - r * HID;
    int sr = r - RELPAD;
    sr = sr < 0 ? 0 : (sr > RELROWS - 1 ? RELROWS - 1 : sr);
    relb[idx] = (__bf16)rp[(long long)sr * HID + c2];
  }
}

// Softmax over the valid diagonal band: row (b,h,q), valid r = 511-q+k.
// Each thread owns exactly 2 of the 512 band elements; exponentials are
// computed once (hardware __expf) and reused for both output layouts.
__global__ __launch_bounds__(256)
void softmax_kernel(const float* __restrict__ SC, __bf16* __restrict__ PB,
                    __bf16* __restrict__ PU, const float* __restrict__ maskv)
{
  __shared__ float red[8];
  const int row = blockIdx.x;
  const int q   = row & (SEQ - 1);
  const int bh  = row >> 9;
  const int b   = bh / NHEADS;
  const float* s  = SC + (long long)bh * SEQ * REL + (long long)q * REL + (511 - q);
  const float* mk = maskv + (long long)b * SEQ;
  const int k0 = threadIdx.x;
  const int k1 = threadIdx.x + 256;
  const float v0 = (s[k0] + mk[k0]) * COEF;
  const float v1 = (s[k1] + mk[k1]) * COEF;
  const float mx = blockReduceMax(fmaxf(v0, v1), red);
  const float e0 = __expf(v0 - mx);
  const float e1 = __expf(v1 - mx);
  const float sum = blockReduceSum(e0 + e1, red);
  const float inv = sum > 0.0f ? 1.0f / sum : 0.0f;
  __bf16* pb = PB + (long long)bh * SEQ * REL + (long long)q * REL;
  __bf16* pu = PU + (long long)bh * SEQ * SEQ + (long long)q * SEQ;
  for (int r = threadIdx.x; r < REL; r += 256) pb[r] = (__bf16)0.0f;
  __syncthreads();
  const float p0 = e0 * inv;
  const float p1 = e1 * inv;
  pb[511 - q + k0] = (__bf16)p0;
  pb[511 - q + k1] = (__bf16)p1;
  pu[k0] = (__bf16)p0;
  pu[k1] = (__bf16)p1;
}

// ---------------------------------------------------------------------------
extern "C" void kernel_launch(void* const* d_in, const int* in_sizes, int n_in,
                              void* d_out, int out_size, void* d_ws, size_t ws_size,
                              hipStream_t stream)
{
  (void)in_sizes; (void)n_in; (void)out_size; (void)ws_size;
  const int*   token_ids   = (const int*)d_in[0];
  const int*   segment_ids = (const int*)d_in[1];
  const int*   input_mask  = (const int*)d_in[2];
  const float* tok_emb     = (const float*)d_in[3];
  const float* seg_emb     = (const float*)d_in[4];
  const float* emb_scale   = (const float*)d_in[5];
  const float* emb_bias    = (const float*)d_in[6];
  const float* Wq = (const float*)d_in[7];
  const float* bq = (const float*)d_in[8];
  const float* Wk = (const float*)d_in[9];
  const float* bk = (const float*)d_in[10];
  const float* Wv = (const float*)d_in[11];
  const float* bv = (const float*)d_in[12];
  const float* relpos = (const float*)d_in[13];
  const float* Wo = (const float*)d_in[14];
  const float* bo = (const float*)d_in[15];
  const float* ln1_s = (const float*)d_in[16];
  const float* ln1_b = (const float*)d_in[17];
  const float* W1 = (const float*)d_in[18];
  const float* b1 = (const float*)d_in[19];
  const float* W2 = (const float*)d_in[20];
  const float* b2 = (const float*)d_in[21];
  const float* ln2_s = (const float*)d_in[22];
  const float* ln2_b = (const float*)d_in[23];
  float* out = (float*)d_out;

  size_t off = 0;
  auto carve = [&](size_t bytes) -> char* {
    char* p = (char*)d_ws + off;
    off += (bytes + 255) & ~(size_t)255;
    return p;
  };
  float*  x    = (float*) carve((size_t)ROWS * HID * 4);
  __bf16* xb   = (__bf16*)carve((size_t)ROWS * HID * 2);
  float*  y    = (float*) carve((size_t)ROWS * HID * 4);
  __bf16* qb   = (__bf16*)carve((size_t)ROWS * HID * 2);
  float*  kf   = (float*) carve((size_t)ROWS * HID * 4);
  float*  vf   = (float*) carve((size_t)ROWS * HID * 4);
  __bf16* relb = (__bf16*)carve((size_t)REL * HID * 2);
  float*  krel = (float*) carve((size_t)REL * HID * 4);
  float*  vrel = (float*) carve((size_t)REL * HID * 4);
  float*  SC   = (float*) carve((size_t)BATCH * NHEADS * SEQ * REL * 4);
  __bf16* PB   = (__bf16*)carve((size_t)BATCH * NHEADS * SEQ * REL * 2);
  __bf16* PU   = (__bf16*)carve((size_t)BATCH * NHEADS * SEQ * SEQ * 2);
  float*  ctx  = (float*) carve((size_t)ROWS * HID * 4);
  __bf16* ctxb = (__bf16*)carve((size_t)ROWS * HID * 2);
  __bf16* midb = (__bf16*)carve((size_t)ROWS * MFF * 2);
  float*  maskv= (float*) carve((size_t)ROWS * 4);

  auto gemm = [&](const __bf16* A, int lda, long long aSb, long long aSh,
                  const float* Bm, int ldb, long long bSb, long long bSh,
                  const float* bias,
                  float* C, int ldc, __bf16* Cb, int ldcb,
                  long long cSb, long long cSh,
                  int M, int N, int K, int flags, int batch, int batchDiv) {
    dim3 grid((N + BN - 1) / BN, (M + BM - 1) / BM, batch);
    wmma_gemm_kernel<<<grid, 256, 0, stream>>>(A, lda, aSb, aSh, Bm, ldb, bSb, bSh,
                                               bias, C, ldc, Cb, ldcb, cSb, cSh,
                                               M, N, K, flags, batchDiv);
  };

  // Embedding + LN + mask
  embed_kernel<<<ROWS, 256, 0, stream>>>(token_ids, segment_ids, tok_emb, seg_emb, y);
  ln_kernel<<<ROWS, 256, 0, stream>>>(y, nullptr, emb_scale, emb_bias, x, xb);
  maskv_kernel<<<(ROWS + 255) / 256, 256, 0, stream>>>(input_mask, maskv, ROWS);

  const long long SCz = (long long)SEQ * REL;        // 524288 per (b,h)
  const long long PUz = (long long)SEQ * SEQ;        // 262144 per (b,h)
  const long long QSb = (long long)SEQ * HID;        // per-batch stride in q/k/v

  for (int i = 0; i < NLAYERS; ++i) {
    const float* Wq_i = Wq + (size_t)i * HID * HID;
    const float* Wk_i = Wk + (size_t)i * HID * HID;
    const float* Wv_i = Wv + (size_t)i * HID * HID;
    const float* Wo_i = Wo + (size_t)i * HID * HID;
    const float* bq_i = bq + (size_t)i * HID;
    const float* bk_i = bk + (size_t)i * HID;
    const float* bv_i = bv + (size_t)i * HID;
    const float* bo_i = bo + (size_t)i * HID;
    const float* W1_i = W1 + (size_t)i * HID * MFF;
    const float* b1_i = b1 + (size_t)i * MFF;
    const float* W2_i = W2 + (size_t)i * MFF * HID;
    const float* b2_i = b2 + (size_t)i * HID;

    // Relative-position embeddings (edge pad) and their K/V projections.
    relpad_kernel<<<(REL * HID + 255) / 256, 256, 0, stream>>>(
        relpos + (size_t)i * RELROWS * HID, relb);
    gemm(relb, HID, 0, 0, Wk_i, HID, 0, 0, bk_i, krel, HID, nullptr, 0, 0, 0,
         REL, HID, HID, 0, 1, 1);
    gemm(relb, HID, 0, 0, Wv_i, HID, 0, 0, bv_i, vrel, HID, nullptr, 0, 0, 0,
         REL, HID, HID, 0, 1, 1);

    // Q (bf16), K (f32), V (f32) projections.
    gemm(xb, HID, 0, 0, Wq_i, HID, 0, 0, bq_i, nullptr, 0, qb, HID, 0, 0,
         ROWS, HID, HID, 0, 1, 1);
    gemm(xb, HID, 0, 0, Wk_i, HID, 0, 0, bk_i, kf, HID, nullptr, 0, 0, 0,
         ROWS, HID, HID, 0, 1, 1);
    gemm(xb, HID, 0, 0, Wv_i, HID, 0, 0, bv_i, vf, HID, nullptr, 0, 0, 0,
         ROWS, HID, HID, 0, 1, 1);

    // Rel scores: SC[z][q][r] = Q . Krel^T    (batched over 48 (b,h))
    gemm(qb, HID, QSb, HDIM,
         krel, HID, 0, HDIM,
         nullptr,
         SC, REL, nullptr, 0,
         (long long)NHEADS * SCz, SCz,
         SEQ, REL, HDIM, FLAG_BT, BATCH * NHEADS, NHEADS);
    // Abs scores accumulated into the skewed layout via ldc=1023, base+511.
    gemm(qb, HID, QSb, HDIM,
         kf, HID, QSb, HDIM,
         nullptr,
         SC + 511, REL - 1, nullptr, 0,
         (long long)NHEADS * SCz, SCz,
         SEQ, SEQ, HDIM, FLAG_BT | FLAG_ACCUM, BATCH * NHEADS, NHEADS);

    softmax_kernel<<<BATCH * NHEADS * SEQ, 256, 0, stream>>>(SC, PB, PU, maskv);

    // rel_ctx = probs(skewed) @ V_rel  (K = 1024)
    gemm(PB, REL, (long long)NHEADS * SCz, SCz,
         vrel, HID, 0, HDIM,
         nullptr,
         ctx, HID, nullptr, 0,
         QSb, HDIM,
         SEQ, HDIM, REL, 0, BATCH * NHEADS, NHEADS);
    // ctx += probs(un-skewed) @ V  (K = 512); also emit bf16 ctx.
    gemm(PU, SEQ, (long long)NHEADS * PUz, PUz,
         vf, HID, QSb, HDIM,
         nullptr,
         ctx, HID, ctxb, HID,
         QSb, HDIM,
         SEQ, HDIM, SEQ, FLAG_ACCUM, BATCH * NHEADS, NHEADS);

    // Output projection + residual LN.
    gemm(ctxb, HID, 0, 0, Wo_i, HID, 0, 0, bo_i, y, HID, nullptr, 0, 0, 0,
         ROWS, HID, HID, 0, 1, 1);
    ln_kernel<<<ROWS, 256, 0, stream>>>(y, x, ln1_s + (size_t)i * HID,
                                        ln1_b + (size_t)i * HID, x, xb);

    // MLP: GELU fused into first GEMM epilogue (bf16 out only).
    gemm(xb, HID, 0, 0, W1_i, MFF, 0, 0, b1_i, nullptr, 0, midb, MFF, 0, 0,
         ROWS, MFF, HID, FLAG_GELU, 1, 1);
    gemm(midb, MFF, 0, 0, W2_i, HID, 0, 0, b2_i, y, HID, nullptr, 0, 0, 0,
         ROWS, HID, MFF, 0, 1, 1);

    float* outf = (i == NLAYERS - 1) ? out : x;
    ln_kernel<<<ROWS, 256, 0, stream>>>(y, x, ln2_s + (size_t)i * HID,
                                        ln2_b + (size_t)i * HID, outf, xb);
  }
}
